// MultiheadAttention_4887672783273
// MI455X (gfx1250) — matrix operand order repaired and checked by reading, not compiled, not run on hardware
//
#include <hip/hip_runtime.h>
#include <hip/hip_bf16.h>
#include <math.h>

// ---------------- CDNA5 WMMA types ----------------
typedef __attribute__((ext_vector_type(16))) __bf16 v16bf;
typedef __attribute__((ext_vector_type(8)))  float  v8f;

constexpr int L_ = 2048;   // sequence length
constexpr int E_ = 1024;   // embed dim
constexpr int H_ = 16;     // heads
constexpr int D_ = 64;     // head dim

struct U128 { unsigned int w[4]; };   // trivially-copyable 16B chunk

union FragBF16 {
  v16bf  v;
  __bf16 e[16];
};

__device__ __forceinline__ v8f zero8() {
  v8f z = {0.f, 0.f, 0.f, 0.f, 0.f, 0.f, 0.f, 0.f};
  return z;
}

__device__ __forceinline__ v8f wmma_bf16(v16bf a, v16bf b, v8f c) {
  // D = A(16x32) * B(32x16) + C, f32 accumulate
  return __builtin_amdgcn_wmma_f32_16x16x32_bf16(
      /*neg_a=*/false, a, /*neg_b=*/false, b,
      /*c_mod=*/(short)0, c, /*reuse_a=*/false, /*reuse_b=*/false);
}

// A-matrix fragment (16x32 bf16). Lane l (l<16): M=l, K = klo..klo+15
// with klo = 0 (lanes 0-15) or 16 (lanes 16-31). 16 contiguous elements.
__device__ __forceinline__ v16bf ld_a_bf16(const __bf16* base, int lda) {
  int lane = threadIdx.x & 31;
  int m    = lane & 15;
  int klo  = lane & 16;                        // 0 or 16
  const __bf16* p = base + (size_t)m * lda + klo;
  FragBF16 f;
  *(U128*)&f.e[0] = *(const U128*)(p);         // K = klo .. klo+7
  *(U128*)&f.e[8] = *(const U128*)(p + 8);     // K = klo+8 .. klo+15
  return f.v;
}

__device__ __forceinline__ v16bf ld_a_f32(const float* base, int lda) {
  int lane = threadIdx.x & 31;
  int m    = lane & 15;
  int klo  = lane & 16;
  const float* p = base + (size_t)m * lda + klo;
  FragBF16 f;
#pragma unroll
  for (int i = 0; i < 8; ++i) f.e[i] = (__bf16)p[i];
#pragma unroll
  for (int i = 0; i < 8; ++i) f.e[8 + i] = (__bf16)p[8 + i];
  return f.v;
}

// B-matrix fragment (32x16 bf16), source stored as Bn[n][k] (row n, contiguous k).
// Lane l (l<16): N=l, K=0..15; lanes 16-31: N=l-16, K=16..31. 16 contiguous elements.
__device__ __forceinline__ v16bf ld_b_bf16(const __bf16* base, int ldb) {
  int lane = threadIdx.x & 31;
  int n    = lane & 15;
  int kb   = lane & 16;                        // 0 or 16
  const __bf16* p = base + (size_t)n * ldb + kb;
  FragBF16 f;
  *(U128*)&f.e[0] = *(const U128*)(p);
  *(U128*)&f.e[8] = *(const U128*)(p + 8);
  return f.v;
}

__device__ __forceinline__ v16bf ld_b_f32(const float* base, int ldb) {
  int lane = threadIdx.x & 31;
  int n    = lane & 15;
  int kb   = lane & 16;
  const float* p = base + (size_t)n * ldb + kb;
  FragBF16 f;
#pragma unroll
  for (int i = 0; i < 16; ++i) f.e[i] = (__bf16)p[i];
  return f.v;
}

// ---------------- GEMM: C[M=2048][N=1024] = A[2048][1024] * W^T + bias ----------------
// W is row-major [N][K] (so x @ W.T has contiguous B-fragments). Block = 8 waves,
// each wave owns a 16x64 tile (4 v8f accumulators). Grid (16, 16).
template <bool ABF16>
__global__ void gemm_xwT(const void* Aptr, const float* W, const float* bias, float* Cout) {
  constexpr int N = 1024, K = 1024;
  int wave = threadIdx.x >> 5;
  int mt   = blockIdx.x * 128 + wave * 16;
  int nb   = blockIdx.y * 64;
  v8f acc[4] = {zero8(), zero8(), zero8(), zero8()};

  for (int k0 = 0; k0 < K; k0 += 32) {
    v16bf a;
    if constexpr (ABF16)
      a = ld_a_bf16((const __bf16*)Aptr + (size_t)mt * K + k0, K);
    else
      a = ld_a_f32((const float*)Aptr + (size_t)mt * K + k0, K);
    __builtin_prefetch(W + (size_t)nb * K + k0 + 64, 0, 1);
#pragma unroll
    for (int j = 0; j < 4; ++j) {
      v16bf b = ld_b_f32(W + (size_t)(nb + 16 * j) * K + k0, K);
      acc[j] = wmma_bf16(a, b, acc[j]);
    }
  }

  int lane = threadIdx.x & 31;
  int mofs = (lane & 16) >> 1;                 // +8 for upper lanes
  int nlo  = lane & 15;
#pragma unroll
  for (int j = 0; j < 4; ++j) {
    int n = nb + 16 * j + nlo;
    float bv = bias[n];
#pragma unroll
    for (int r = 0; r < 8; ++r) {
      int m = mt + r + mofs;
      Cout[(size_t)m * N + n] = acc[j][r] + bv;
    }
  }
}

// ---------------- xPos rotary + head split + V transpose ----------------
// proj = [q | k | v] f32 [L][E]. Emits qh/kh bf16 [H][L][D], vT bf16 [H][D][L].
__global__ void xpos_qkv(const float* __restrict__ proj,
                         __bf16* __restrict__ qh, __bf16* __restrict__ kh,
                         __bf16* __restrict__ vT) {
  int idx = blockIdx.x * 256 + threadIdx.x;    // H*L*32 threads, one per (h,t,pair)
  int j = idx & 31;
  int t = (idx >> 5) & (L_ - 1);
  int h = idx >> 16;
  int e = h * D_ + 2 * j;

  const float* qp = proj;
  const float* kp = proj + (size_t)L_ * E_;
  const float* vp = proj + 2 * (size_t)L_ * E_;
  size_t ro = (size_t)t * E_ + e;
  float q1 = qp[ro], q2 = qp[ro + 1];
  float k1 = kp[ro], k2 = kp[ro + 1];
  float v1 = vp[ro], v2 = vp[ro + 1];

  float base_scale = (2.0f * (float)j + 0.4f * (float)D_) / (1.4f * (float)D_);
  float pos        = (float)t - (float)(L_ / 2);
  float scale      = powf(base_scale, pos / 512.0f);
  float inv_freq   = powf(10000.0f, -(float)j / 32.0f);
  float ang = (float)t * inv_freq;
  float sn = sinf(ang), cs = cosf(ang);

  float cq = cs * scale, sq = sn * scale;      // q: upscale
  float is = 1.0f / scale;
  float ck = cs * is, sk = sn * is;            // k: downscale
  const float qs = 0.125f;                     // D^-0.5

  float qo1 = (q1 * cq - q2 * sq) * qs;
  float qo2 = (q2 * cq + q1 * sq) * qs;
  float ko1 = k1 * ck - k2 * sk;
  float ko2 = k2 * ck + k1 * sk;

  size_t hb = ((size_t)h * L_ + t) * D_ + 2 * j;
  qh[hb] = (__bf16)qo1; qh[hb + 1] = (__bf16)qo2;
  kh[hb] = (__bf16)ko1; kh[hb + 1] = (__bf16)ko2;

  size_t vb = ((size_t)h * D_ + 2 * j) * L_ + t;
  vT[vb]      = (__bf16)v1;
  vT[vb + L_] = (__bf16)v2;
}

// ---------------- logits: aw[h][t][s] = q.kT + mask + rel_pos ----------------
// Grid (16, 32, 16): x = t-tiles of 128, y = s-tiles of 64, z = head.
__global__ void qk_kernel(const __bf16* __restrict__ qh, const __bf16* __restrict__ kh,
                          const float* __restrict__ mask, const float* __restrict__ relpos,
                          float* __restrict__ aw) {
  int h    = blockIdx.z;
  int wave = threadIdx.x >> 5;
  int tt   = blockIdx.x * 128 + wave * 16;
  int sb   = blockIdx.y * 64;
  const __bf16* qbase = qh + ((size_t)h * L_ + tt) * D_;
  const __bf16* kbase = kh + (size_t)h * L_ * D_;

  v8f acc[4] = {zero8(), zero8(), zero8(), zero8()};
#pragma unroll
  for (int d0 = 0; d0 < D_; d0 += 32) {
    v16bf a = ld_a_bf16(qbase + d0, D_);
#pragma unroll
    for (int j = 0; j < 4; ++j) {
      v16bf b = ld_b_bf16(kbase + (size_t)(sb + 16 * j) * D_ + d0, D_);
      acc[j] = wmma_bf16(a, b, acc[j]);
    }
  }

  int lane = threadIdx.x & 31;
  int mofs = (lane & 16) >> 1;
  int nlo  = lane & 15;
#pragma unroll
  for (int j = 0; j < 4; ++j) {
    int s = sb + 16 * j + nlo;
#pragma unroll
    for (int r = 0; r < 8; ++r) {
      int t = tt + r + mofs;
      size_t o = ((size_t)h * L_ + t) * L_ + s;
      aw[o] = acc[j][r] + mask[(size_t)t * L_ + s] + relpos[o];
    }
  }
}

// ---------------- in-place stable softmax over rows of aw ----------------
__global__ void softmax_rows(float* __restrict__ aw) {
  float* p = aw + (size_t)blockIdx.x * L_;
  __shared__ float red[256];
  int tid = threadIdx.x;

  float m = -3.0e38f;
  for (int i = tid; i < L_; i += 256) m = fmaxf(m, p[i]);
  red[tid] = m;
  __syncthreads();
  for (int o = 128; o > 0; o >>= 1) {
    if (tid < o) red[tid] = fmaxf(red[tid], red[tid + o]);
    __syncthreads();
  }
  m = red[0];
  __syncthreads();

  float sum = 0.f;
  for (int i = tid; i < L_; i += 256) { float ex = __expf(p[i] - m); p[i] = ex; sum += ex; }
  red[tid] = sum;
  __syncthreads();
  for (int o = 128; o > 0; o >>= 1) {
    if (tid < o) red[tid] += red[tid + o];
    __syncthreads();
  }
  float inv = 1.0f / red[0];
  for (int i = tid; i < L_; i += 256) p[i] *= inv;
}

// ---------------- attn[t][h*64+d] = sum_s aw[h][t][s] * v[h][s][d] ----------------
// vT is [H][D][L] so B-fragments (k=s contiguous) are contiguous. Grid (16, 16): x = t-tiles, y = head.
__global__ void pv_kernel(const float* __restrict__ aw, const __bf16* __restrict__ vT,
                          float* __restrict__ attn) {
  int h    = blockIdx.y;
  int wave = threadIdx.x >> 5;
  int tt   = blockIdx.x * 128 + wave * 16;
  const float*  abase = aw + ((size_t)h * L_ + tt) * L_;
  const __bf16* vbase = vT + (size_t)h * D_ * L_;

  v8f acc[4] = {zero8(), zero8(), zero8(), zero8()};
  for (int s0 = 0; s0 < L_; s0 += 32) {
    v16bf a = ld_a_f32(abase + s0, L_);
#pragma unroll
    for (int j = 0; j < 4; ++j) {
      v16bf b = ld_b_bf16(vbase + (size_t)(16 * j) * L_ + s0, L_);
      acc[j] = wmma_bf16(a, b, acc[j]);
    }
  }

  int lane = threadIdx.x & 31;
  int mofs = (lane & 16) >> 1;
  int nlo  = lane & 15;
#pragma unroll
  for (int j = 0; j < 4; ++j) {
    int d = 16 * j + nlo;
#pragma unroll
    for (int r = 0; r < 8; ++r) {
      int t = tt + r + mofs;
      attn[(size_t)t * E_ + h * D_ + d] = acc[j][r];
    }
  }
}

// ---------------- layernorm over rows, emit bf16 for final GEMM ----------------
__global__ void layernorm_rows(const float* __restrict__ x, const float* __restrict__ g,
                               const float* __restrict__ b, __bf16* __restrict__ xn) {
  int t = blockIdx.x;
  const float* p = x + (size_t)t * E_;
  __shared__ float s1[256], s2[256];
  int tid = threadIdx.x;

  float sum = 0.f, sq = 0.f;
  for (int i = tid; i < E_; i += 256) { float v = p[i]; sum += v; sq += v * v; }
  s1[tid] = sum; s2[tid] = sq;
  __syncthreads();
  for (int o = 128; o > 0; o >>= 1) {
    if (tid < o) { s1[tid] += s1[tid + o]; s2[tid] += s2[tid + o]; }
    __syncthreads();
  }
  float mean = s1[0] * (1.0f / E_);
  float var  = s2[0] * (1.0f / E_) - mean * mean;
  float rstd = rsqrtf(var + 1e-5f);
  for (int i = tid; i < E_; i += 256)
    xn[(size_t)t * E_ + i] = (__bf16)((p[i] - mean) * rstd * g[i] + b[i]);
}

// ---------------- host-side orchestration ----------------
extern "C" void kernel_launch(void* const* d_in, const int* in_sizes, int n_in,
                              void* d_out, int out_size, void* d_ws, size_t ws_size,
                              hipStream_t stream) {
  const float* query   = (const float*)d_in[0];
  const float* rel_pos = (const float*)d_in[1];
  const float* mask    = (const float*)d_in[2];
  const float* wq = (const float*)d_in[3];
  const float* bq = (const float*)d_in[4];
  const float* wk = (const float*)d_in[5];
  const float* bk = (const float*)d_in[6];
  const float* wv = (const float*)d_in[7];
  const float* bv = (const float*)d_in[8];
  const float* wo = (const float*)d_in[9];
  const float* bo = (const float*)d_in[10];
  const float* lng = (const float*)d_in[11];
  const float* lnb = (const float*)d_in[12];

  float* out = (float*)d_out;                       // [L][E]
  float* aw  = out + (size_t)L_ * E_;               // [H][L][L]

  // workspace layout (~36 MB):
  //   [0, 24MB)  : proj f32 [3][L][E]   -- dead after xpos_qkv; region reused:
  //       [0, 8MB)   attn f32 [L][E]    (pv_kernel output)
  //       [8MB,12MB) xn  bf16 [L][E]    (layernorm output)
  //   [24MB, 36MB): qh, kh bf16 [H][L][D]; vT bf16 [H][D][L]
  char*   ws   = (char*)d_ws;
  float*  proj = (float*)ws;
  float*  attn = (float*)ws;
  __bf16* xn   = (__bf16*)(ws + (size_t)L_ * E_ * 4);
  __bf16* qh   = (__bf16*)(ws + (size_t)3 * L_ * E_ * 4);
  __bf16* kh   = qh + (size_t)H_ * L_ * D_;
  __bf16* vT   = kh + (size_t)H_ * L_ * D_;

  dim3 blk(256);

  // Q/K/V projections (bf16 WMMA, f32 accumulate)
  gemm_xwT<false><<<dim3(16, 16), blk, 0, stream>>>(query, wq, bq, proj);
  gemm_xwT<false><<<dim3(16, 16), blk, 0, stream>>>(query, wk, bk, proj + (size_t)L_ * E_);
  gemm_xwT<false><<<dim3(16, 16), blk, 0, stream>>>(query, wv, bv, proj + (size_t)2 * L_ * E_);

  // xPos rotary + head split + V transpose
  xpos_qkv<<<(H_ * L_ * 32) / 256, blk, 0, stream>>>(proj, qh, kh, vT);

  // logits + mask + rel_pos -> aw region of d_out
  qk_kernel<<<dim3(16, 32, 16), blk, 0, stream>>>(qh, kh, mask, rel_pos, aw);

  // softmax in place
  softmax_rows<<<H_ * L_, blk, 0, stream>>>(aw);

  // attn = aw @ v
  pv_kernel<<<dim3(16, 16), blk, 0, stream>>>(aw, vT, attn);

  // layernorm -> bf16
  layernorm_rows<<<L_, blk, 0, stream>>>(attn, lng, lnb, xn);

  // output projection
  gemm_xwT<true><<<dim3(16, 16), blk, 0, stream>>>(xn, wo, bo, out);
}
